// TopoGrad_60576218743167
// MI455X (gfx1250) — matrix operand order repaired and checked by reading, not compiled, not run on hardware
//
#include <hip/hip_runtime.h>
#include <stdint.h>

#define NPTS    8192
#define DIM     128
#define KRIPS   15
#define ROWS_WG 64
#define CCHUNK  64
#define BSTRIDE (DIM + 8)   // uint16 elements -> 272 B row stride (16B aligned, bank-skewed)

typedef __bf16 v16bf __attribute__((ext_vector_type(16)));
typedef float  v8f   __attribute__((ext_vector_type(8)));

union FragU { v16bf v; uint4 q[2]; };

// A-matrix 16x32 bf16 fragment (ISA 7.12.2): lanes 0-15 row M=lane, K = kbase+0..7 & kbase+16..23;
// lanes 16-31 same rows, K = kbase+8..15 & kbase+24..31. Two 16B contiguous chunks per lane.
__device__ __forceinline__ v16bf load_a_frag(const uint16_t* __restrict__ x,
                                             int rowBase, int kbase, int lane) {
  int lo = lane & 15;
  int koff = (lane >> 4) ? 8 : 0;
  const uint16_t* p = x + (size_t)(rowBase + lo) * DIM + kbase + koff;
  FragU u;
  u.q[0] = *(const uint4*)(p);
  u.q[1] = *(const uint4*)(p + 16);
  return u.v;
}

// B-matrix 32x16 bf16 fragment: lane holds column N = colBase + (lane&15);
// lanes 0-15 cover K=kbase+0..7 & kbase+16..23, lanes 16-31 K=kbase+8..15 & kbase+24..31.
// Since B[k][n] = X[n][k], this is two 16B contiguous loads per lane from row-major X.
__device__ __forceinline__ v16bf load_b_frag(const uint16_t* __restrict__ x,
                                             int colBase, int kbase, int lane) {
  int lo = lane & 15;
  int koff = (lane >> 4) ? 8 : 0;
  const uint16_t* p = x + (size_t)(colBase + lo) * DIM + kbase + koff;
  FragU u;
  u.q[0] = *(const uint4*)(p);
  u.q[1] = *(const uint4*)(p + 16);
  return u.v;
}

// Async Global -> LDS copy, 16B per lane (GLOBAL_LOAD_ASYNC_TO_LDS_B128, ASYNCcnt).
__device__ __forceinline__ void async_copy_b128(uint32_t ldsOff, const uint16_t* gptr) {
  asm volatile("global_load_async_to_lds_b128 %0, %1, off"
               :: "v"(ldsOff), "v"(gptr)
               : "memory");
}

__device__ __forceinline__ void wait_async_all() {
  asm volatile("s_wait_asynccnt 0x0" ::: "memory");
}

// ---------------- Kernel 1: f32 -> bf16 (RNE) + row squared-norms ----------------
__global__ void __launch_bounds__(32)
bf16_norms(const float* __restrict__ x, uint16_t* __restrict__ xb, float* __restrict__ sq) {
  int r = blockIdx.x, lane = threadIdx.x;
  float4 f = ((const float4*)(x + (size_t)r * DIM))[lane];
  float vals[4] = {f.x, f.y, f.z, f.w};
  ushort4 h;
  unsigned short* hp = &h.x;
  float ss = 0.f;
#pragma unroll
  for (int i = 0; i < 4; ++i) {
    uint32_t u = __float_as_uint(vals[i]);
    uint32_t rr = (u + 0x7FFFu + ((u >> 16) & 1u)) >> 16;
    hp[i] = (unsigned short)rr;
    float b = __uint_as_float(rr << 16);
    ss += b * b;   // norms consistent with bf16-quantized points
  }
  ((ushort4*)(xb + (size_t)r * DIM))[lane] = h;
#pragma unroll
  for (int m = 16; m >= 1; m >>= 1) ss += __shfl_xor(ss, m, 32);
  if (lane == 0) sq[r] = ss;
}

// ---------------- Kernel 2: KDE density via WMMA Gram tiles ----------------
// dens[i] = sum_j exp(-2 * d2_ij); contributions outside the top-50 are ~exp(-500) == 0,
// so the full sum equals the reference's top-50 sum to f32 precision.
__global__ void __launch_bounds__(256)
kde_density(const uint16_t* __restrict__ xb, const float* __restrict__ sq,
            float* __restrict__ dens) {
  __shared__ float part[8][16];
  int wave = threadIdx.x >> 5, lane = threadIdx.x & 31;
  int lo = lane & 15, hi = lane >> 4;
  int rowBase = blockIdx.x * 16;

  v16bf a[4];
#pragma unroll
  for (int k = 0; k < 4; ++k) a[k] = load_a_frag(xb, rowBase, k * 32, lane);
  float sqA[8];
#pragma unroll
  for (int v = 0; v < 8; ++v) sqA[v] = sq[rowBase + v + hi * 8];

  float densAcc[8] = {0.f, 0.f, 0.f, 0.f, 0.f, 0.f, 0.f, 0.f};

  for (int ct = wave; ct < NPTS / 16; ct += 8) {
    int colBase = ct * 16;
    v8f acc = {0.f, 0.f, 0.f, 0.f, 0.f, 0.f, 0.f, 0.f};
#pragma unroll
    for (int k = 0; k < 4; ++k) {
      v16bf b = load_b_frag(xb, colBase, k * 32, lane);
      acc = __builtin_amdgcn_wmma_f32_16x16x32_bf16(false, a[k], false, b,
                                                    (short)0, acc, false, false);
    }
    float sqB = sq[colBase + lo];
#pragma unroll
    for (int v = 0; v < 8; ++v) {
      float d2 = sqA[v] + sqB - 2.f * acc[v];
      densAcc[v] += __expf(-2.f * d2);   // exp(-d2/SCALE), SCALE=0.5
    }
  }
  // reduce across the 16 lanes that share each row (halves independent: masks < 16)
#pragma unroll
  for (int v = 0; v < 8; ++v) {
#pragma unroll
    for (int m = 8; m >= 1; m >>= 1) densAcc[v] += __shfl_xor(densAcc[v], m, 32);
  }
  if (lo == 0) {
#pragma unroll
    for (int v = 0; v < 8; ++v) part[wave][v + hi * 8] = densAcc[v];
  }
  __syncthreads();
  if (threadIdx.x < 16) {   // fixed-order cross-wave sum -> deterministic
    float s = 0.f;
#pragma unroll
    for (int w = 0; w < 8; ++w) s += part[w][threadIdx.x];
    dens[rowBase + threadIdx.x] = s;
  }
}

// ---------------- Kernel 3: bitonic argsort of 8192 densities (1 WG) ----------------
__global__ void __launch_bounds__(1024)
sort_density(const float* __restrict__ dens, float* __restrict__ densSorted,
             int* __restrict__ order) {
  __shared__ float key[NPTS];
  __shared__ int   idx[NPTS];
  for (int i = threadIdx.x; i < NPTS; i += 1024) { key[i] = dens[i]; idx[i] = i; }
  __syncthreads();
  for (int k = 2; k <= NPTS; k <<= 1) {
    for (int j = k >> 1; j > 0; j >>= 1) {
      for (int i = threadIdx.x; i < NPTS; i += 1024) {
        int p = i ^ j;
        if (p > i) {
          float ka = key[i], kb = key[p];
          int   ia = idx[i], ib = idx[p];
          bool agtb = (ka > kb) || (ka == kb && ia > ib);  // lexicographic, stable ties
          bool asc  = ((i & k) == 0);
          if (agtb == asc) { key[i] = kb; key[p] = ka; idx[i] = ib; idx[p] = ia; }
        }
      }
      __syncthreads();
    }
  }
  // ascending sort => max density is the last key; normalization preserves order
  float inv = 1.f / key[NPTS - 1];
  for (int i = threadIdx.x; i < NPTS; i += 1024) {
    densSorted[i] = key[i] * inv;
    order[i] = idx[i];
  }
}

// ---------------- Kernel 4: gather points/norms in density order ----------------
__global__ void __launch_bounds__(32)
gather_sorted(const uint16_t* __restrict__ xb, const float* __restrict__ sq,
              const int* __restrict__ order, uint16_t* __restrict__ xs,
              float* __restrict__ sqs) {
  int r = blockIdx.x, lane = threadIdx.x;
  int src = order[r];
  ((uint2*)(xs + (size_t)r * DIM))[lane] = ((const uint2*)(xb + (size_t)src * DIM))[lane];
  if (lane == 0) sqs[r] = sq[src];
}

// ---------------- Kernel 5: WMMA distance tiles (async-LDS-staged B) + top-15 ----------------
__global__ void __launch_bounds__(128)
rips_knn(const uint16_t* __restrict__ xs, const float* __restrict__ sqs,
         int* __restrict__ outIdx) {
  __shared__ float    dist[ROWS_WG][CCHUNK + 1];   // +1 pad: conflict-free row scans
  __shared__ uint16_t Bs[CCHUNK][BSTRIDE];         // staged B rows, 272B stride
  int tid = threadIdx.x, wave = tid >> 5, lane = tid & 31;
  int lo = lane & 15, hi = lane >> 4;
  int wgRow = blockIdx.x * ROWS_WG;
  int rowBase = wgRow + wave * 16;

  v16bf a[4];
#pragma unroll
  for (int k = 0; k < 4; ++k) a[k] = load_a_frag(xs, rowBase, k * 32, lane);
  float sqA[8];
#pragma unroll
  for (int v = 0; v < 8; ++v) sqA[v] = sqs[rowBase + v + hi * 8];

  float bd[KRIPS]; int bi[KRIPS];
#pragma unroll
  for (int q = 0; q < KRIPS; ++q) { bd[q] = 3.4e38f; bi[q] = 0x7FFFFFFF; }

  for (int cb = 0; cb < NPTS; cb += CCHUNK) {
    // Stage this chunk's 64 B-rows (16KB) into LDS once per WG: 8 async b128 per thread.
    // Issued after the previous iteration's trailing barrier, so no wave still reads Bs.
#pragma unroll
    for (int it = 0; it < (CCHUNK * DIM * 2) / (16 * 128); ++it) {   // 8
      int li = it * 128 + tid;          // 16B chunk index within the 16KB stage
      int row = li >> 4;                // 16 x 16B chunks per 256B row
      int inRow = (li & 15) * 16;       // byte offset within row
      uint32_t ldsOff = (uint32_t)(uintptr_t)(&Bs[row][0]) + (uint32_t)inRow;
      async_copy_b128(ldsOff, xs + (size_t)(cb + row) * DIM + (inRow >> 1));
    }
    // Warm L2 for the next chunk while this one computes (global_prefetch_b8).
    if (cb + CCHUNK < NPTS)
      __builtin_prefetch(xs + (size_t)(cb + CCHUNK + (tid >> 1)) * DIM + (tid & 1) * 64, 0, 1);
    wait_async_all();
    __syncthreads();

    for (int ct = 0; ct < CCHUNK / 16; ++ct) {
      int colBase = cb + ct * 16;
      v8f acc = {0.f, 0.f, 0.f, 0.f, 0.f, 0.f, 0.f, 0.f};
#pragma unroll
      for (int k = 0; k < 4; ++k) {
        // B fragment from LDS: lane -> column (ct*16 + lo), two 16B chunks at koff, koff+16 (ds_load_b128 x2)
        int koff = hi ? 8 : 0;
        const uint16_t* p = &Bs[ct * 16 + lo][k * 32 + koff];
        FragU u;
        u.q[0] = *(const uint4*)(p);
        u.q[1] = *(const uint4*)(p + 16);
        acc = __builtin_amdgcn_wmma_f32_16x16x32_bf16(false, a[k], false, u.v,
                                                      (short)0, acc, false, false);
      }
      float sqB = sqs[colBase + lo];
#pragma unroll
      for (int v = 0; v < 8; ++v)
        dist[wave * 16 + v + hi * 8][ct * 16 + lo] = sqA[v] + sqB - 2.f * acc[v];
    }
    __syncthreads();

    if (tid < ROWS_WG) {   // one thread per row: online top-15 by (d2, idx)
      for (int c = 0; c < CCHUNK; ++c) {
        float d = dist[tid][c];
        int gi = cb + c;
        if (d < bd[KRIPS - 1] || (d == bd[KRIPS - 1] && gi < bi[KRIPS - 1])) {
          float cd = d; int ci = gi;
#pragma unroll
          for (int q = 0; q < KRIPS; ++q) {
            bool lt = (cd < bd[q]) || (cd == bd[q] && ci < bi[q]);
            if (lt) {
              float td = bd[q]; int ti = bi[q];
              bd[q] = cd; bi[q] = ci; cd = td; ci = ti;
            }
          }
        }
      }
    }
    __syncthreads();
  }

  if (tid < ROWS_WG) {
    int row = wgRow + tid;
#pragma unroll
    for (int q = 0; q < KRIPS; ++q) outIdx[(size_t)row * KRIPS + q] = bi[q];
  }
}

// ---------------- launcher ----------------
extern "C" void kernel_launch(void* const* d_in, const int* in_sizes, int n_in,
                              void* d_out, int out_size, void* d_ws, size_t ws_size,
                              hipStream_t stream) {
  const float* x = (const float*)d_in[0];

  // d_out: [dens_sorted: 8192 f32][rips_idxs: 8192*15 i32] (bit-compatible regions)
  float* densSorted = (float*)d_out;
  int*   ripsIdx    = (int*)(densSorted + NPTS);

  // workspace layout (~4.3 MB)
  char* w = (char*)d_ws;
  uint16_t* xb  = (uint16_t*)w;                                   // 2 MB bf16 points
  uint16_t* xs  = (uint16_t*)(w + (size_t)NPTS * DIM * 2);        // 2 MB sorted bf16
  float*    sq  = (float*)(w + 2 * (size_t)NPTS * DIM * 2);       // 32 KB norms
  float*    sqs = sq + NPTS;                                      // 32 KB sorted norms
  float*    dens = sqs + NPTS;                                    // 32 KB densities
  int*      order = (int*)(dens + NPTS);                          // 32 KB argsort

  bf16_norms   <<<NPTS, 32, 0, stream>>>(x, xb, sq);
  kde_density  <<<NPTS / 16, 256, 0, stream>>>(xb, sq, dens);
  sort_density <<<1, 1024, 0, stream>>>(dens, densSorted, order);
  gather_sorted<<<NPTS, 32, 0, stream>>>(xb, sq, order, xs, sqs);
  rips_knn     <<<NPTS / ROWS_WG, 128, 0, stream>>>(xs, sqs, ripsIdx);
}
